// CQAttention_67104569032841
// MI455X (gfx1250) — matrix operand order repaired and checked by reading, not compiled, not run on hardware
//
#include <hip/hip_runtime.h>
#include <hip/hip_bf16.h>

// ---------------- problem constants ----------------
#define BATCH 64
#define CL    2048
#define QL    512
#define DIM   128
#define NEG_INF (-1e30f)

typedef __attribute__((ext_vector_type(16))) _Float16 v16h;
typedef __attribute__((ext_vector_type(8)))  _Float16 v8h;
typedef __attribute__((ext_vector_type(8)))  float    v8f;
typedef __attribute__((ext_vector_type(4)))  unsigned int u32x4;
typedef __attribute__((ext_vector_type(8)))  int          i32x8;
typedef __attribute__((ext_vector_type(4)))  int          i32x4;

__device__ __forceinline__ v8f vzero8() {
    return (v8f){0.f,0.f,0.f,0.f,0.f,0.f,0.f,0.f};
}

// WMMA: D = A(16x32 f16) * B(32x16 f16) + C(16x16 f32)
__device__ __forceinline__ v8f wmma_f16(v16h a, v16h b, v8f c) {
    return __builtin_amdgcn_wmma_f32_16x16x32_f16(
        /*neg_a=*/false, a, /*neg_b=*/false, b,
        /*c_mod=*/(short)0, c, /*reuse_a=*/false, /*reuse_b=*/false);
}

// A-fragment (16x32, M x K) from row-major [row][k], leading dim ld (halfs).
__device__ __forceinline__ v16h load_a_frag(const _Float16* base, int ld) {
    const int lane = threadIdx.x & 31;
    const int row  = lane & 15;
    const int kb   = (lane >> 4) << 3;      // 0 or 8
    const _Float16* p0 = base + row * ld + kb;
    v8h lo = *(const v8h*)(p0);
    v8h hi = *(const v8h*)(p0 + 16);
    v16h r;
#pragma unroll
    for (int i = 0; i < 8; ++i) { r[i] = lo[i]; r[i + 8] = hi[i]; }
    return r;
}

// B-fragment (32x16, K x N) read from B^T stored row-major as [n][k], ld in halfs.
__device__ __forceinline__ v16h load_b_frag(const _Float16* base, int ld) {
    const int lane = threadIdx.x & 31;
    const int n  = lane & 15;
    const int kb = (lane >> 4) << 3;        // 0 or 8
    const _Float16* p0 = base + n * ld + kb;
    v8h lo = *(const v8h*)(p0);
    v8h hi = *(const v8h*)(p0 + 16);
    v16h r;
#pragma unroll
    for (int i = 0; i < 8; ++i) { r[i] = lo[i]; r[i + 8] = hi[i]; }
    return r;
}

// ---- Tensor Data Mover: async 2D tile load  global -> LDS (ISA ch.8) ----
// Loads a 64-row x 1024B tile of f16 (row pitch 1024B in memory) into LDS,
// inserting 16B of padding after every 1024B so the LDS row stride is 1040B
// (= 520 halfs, bank-conflict-free). Tracked by TENSORcnt.
__device__ __forceinline__ void tdm_load_s2_tile(const _Float16* gsrc,
                                                 unsigned lds_byte_addr) {
    unsigned long long ga = (unsigned long long)(uintptr_t)gsrc;
    u32x4 g0;
    g0[0] = 1u;                                   // count=1 (valid user D#)
    g0[1] = lds_byte_addr;                        // lds_addr
    g0[2] = (unsigned)(ga & 0xffffffffu);         // global_addr[31:0]
    g0[3] = (unsigned)((ga >> 32) & 0x1ffffffu)   // global_addr[56:32]
          | (2u << 30);                           // type = 2 ("image")
    i32x8 g1;
    g1[0] = (3 << 16)     // data_size = 8 bytes
          | (1 << 20)     // pad_enable
          | (7 << 22)     // pad_interval code 7 = 256 DWORDs (1024B)
          | (3 << 25);    // pad_amount  code 3 = 4 DWORDs (16B)
    g1[1] = (int)(128u << 16);   // tensor_dim0 = 128 units (1024B rows)
    g1[2] = (int)(2048u << 16);  // tensor_dim1 = 2048 rows
    g1[3] = (int)(128u << 16);   // tile_dim0 = 128 units
    g1[4] = 64;                  // tile_dim1 = 64 rows, tile_dim2 = 0
    g1[5] = 128;                 // tensor_dim0_stride = 128 units
    g1[6] = 0;
    g1[7] = 0;
    i32x4 z4 = (i32x4){0, 0, 0, 0};
    i32x8 z8 = (i32x8){0, 0, 0, 0, 0, 0, 0, 0};
    // clang-23 / therock-10.0 6-arg form: (g0, g1, g2, g3, ext, cpol)
    __builtin_amdgcn_tensor_load_to_lds(g0, g1, z4, z4, z8, 0);
}

// ---------------- prep kernels ----------------
__global__ __launch_bounds__(256) void k_prep_c(
    const float* __restrict__ C, const float* __restrict__ w4c,
    const float* __restrict__ w4mul,
    _Float16* __restrict__ Cwh, _Float16* __restrict__ ChT,
    float* __restrict__ s0) {
    int idx = blockIdx.x * 256 + threadIdx.x;       // over BATCH*CL
    if (idx >= BATCH * CL) return;
    int b = idx / CL, c = idx % CL;
    const float* row = C + (size_t)idx * DIM;
    float acc = 0.f;
#pragma unroll 4
    for (int d = 0; d < DIM; ++d) {
        float v = row[d];
        acc += v * w4c[d];
        Cwh[(size_t)idx * DIM + d] = (_Float16)(v * w4mul[d]);
        ChT[((size_t)b * DIM + d) * CL + c] = (_Float16)v;
    }
    s0[idx] = acc;
}

__global__ __launch_bounds__(256) void k_prep_q(
    const float* __restrict__ Q, const float* __restrict__ w4q,
    _Float16* __restrict__ Qh, _Float16* __restrict__ QhT,
    float* __restrict__ s1) {
    int idx = blockIdx.x * 256 + threadIdx.x;       // over BATCH*QL
    if (idx >= BATCH * QL) return;
    int b = idx / QL, q = idx % QL;
    const float* row = Q + (size_t)idx * DIM;
    float acc = 0.f;
#pragma unroll 4
    for (int d = 0; d < DIM; ++d) {
        float v = row[d];
        acc += v * w4q[d];
        Qh[(size_t)idx * DIM + d] = (_Float16)v;
        QhT[((size_t)b * DIM + d) * QL + q] = (_Float16)v;
    }
    s1[idx] = acc;
}

// ---------------- shared S-tile compute (32 x 512) ----------------
#define LSLD 513   // padded f32 row stride

__device__ __forceinline__ void compute_s_tile32(
    float* ls, int b, int c0,
    const _Float16* __restrict__ Cwh, const _Float16* __restrict__ Qh,
    const float* __restrict__ s0, const float* __restrict__ s1, float bz) {
    const int w    = threadIdx.x >> 5;
    const int lane = threadIdx.x & 31;
    const int n    = lane & 15;
    const int mb   = (lane >> 4) * 8;
    const int m    = w & 1;          // 2 M-tiles (32 rows)
    const int ng   = w >> 1;         // 4 groups x 128 q-cols

    const _Float16* Arow = Cwh + (size_t)(b * CL + c0 + m * 16) * DIM;
    v16h af[4];
#pragma unroll
    for (int kc = 0; kc < 4; ++kc) af[kc] = load_a_frag(Arow + kc * 32, DIM);

    float s0r[8];
#pragma unroll
    for (int r = 0; r < 8; ++r) s0r[r] = s0[b * CL + c0 + m * 16 + mb + r];

    for (int nt = 0; nt < 8; ++nt) {
        int q0 = ng * 128 + nt * 16;
        v8f acc = vzero8();
#pragma unroll
        for (int kc = 0; kc < 4; ++kc) {
            v16h bf = load_b_frag(Qh + (size_t)(b * QL + q0) * DIM + kc * 32, DIM);
            acc = wmma_f16(af[kc], bf, acc);
        }
        float s1v = s1[b * QL + q0 + n];
#pragma unroll
        for (int r = 0; r < 8; ++r)
            ls[(m * 16 + mb + r) * LSLD + q0 + n] = acc[r] + s0r[r] + s1v + bz;
    }
}

// ---------------- softmax statistics ----------------
__global__ __launch_bounds__(256) void k_stats(
    const _Float16* __restrict__ Cwh, const _Float16* __restrict__ Qh,
    const float* __restrict__ s0, const float* __restrict__ s1,
    const float* __restrict__ bias,
    const float* __restrict__ q_mask, const float* __restrict__ c_mask,
    float* __restrict__ rmax, float* __restrict__ rsum,
    float* __restrict__ pmax, float* __restrict__ psum) {
    __shared__ float ls[32 * LSLD];
    const int b  = blockIdx.x / (CL / 32);
    const int ct = blockIdx.x % (CL / 32);
    const int c0 = ct * 32;

    compute_s_tile32(ls, b, c0, Cwh, Qh, s0, s1, bias[0]);
    __syncthreads();

    if (threadIdx.x < 32) {
        int r = threadIdx.x;
        float mx = -3.4e38f;
        for (int q = 0; q < QL; ++q) {
            float qm = q_mask[b * QL + q];
            float sv = ls[r * LSLD + q] * qm + NEG_INF * (1.f - qm);
            mx = fmaxf(mx, sv);
        }
        float sm = 0.f;
        for (int q = 0; q < QL; ++q) {
            float qm = q_mask[b * QL + q];
            float sv = ls[r * LSLD + q] * qm + NEG_INF * (1.f - qm);
            sm += __expf(sv - mx);
        }
        rmax[b * CL + c0 + r] = mx;
        rsum[b * CL + c0 + r] = sm;
    }

    for (int q = threadIdx.x; q < QL; q += 256) {
        float mx = -3.4e38f;
        for (int r = 0; r < 32; ++r) {
            float cm = c_mask[b * CL + c0 + r];
            float sv = ls[r * LSLD + q] * cm + NEG_INF * (1.f - cm);
            mx = fmaxf(mx, sv);
        }
        float sm = 0.f;
        for (int r = 0; r < 32; ++r) {
            float cm = c_mask[b * CL + c0 + r];
            float sv = ls[r * LSLD + q] * cm + NEG_INF * (1.f - cm);
            sm += __expf(sv - mx);
        }
        size_t pi = ((size_t)b * (CL / 32) + ct) * QL + q;
        pmax[pi] = mx;
        psum[pi] = sm;
    }
}

__global__ __launch_bounds__(256) void k_colred(
    const float* __restrict__ pmax, const float* __restrict__ psum,
    float* __restrict__ cmax, float* __restrict__ csum) {
    int idx = blockIdx.x * 256 + threadIdx.x;       // over BATCH*QL
    if (idx >= BATCH * QL) return;
    int b = idx / QL, q = idx % QL;
    float mx = -3.4e38f;
    for (int t = 0; t < CL / 32; ++t)
        mx = fmaxf(mx, pmax[((size_t)b * (CL / 32) + t) * QL + q]);
    float sm = 0.f;
    for (int t = 0; t < CL / 32; ++t) {
        size_t pi = ((size_t)b * (CL / 32) + t) * QL + q;
        sm += psum[pi] * __expf(pmax[pi] - mx);
    }
    cmax[idx] = mx;
    csum[idx] = sm;
}

// ---------------- materialize S2 (f16) ----------------
__global__ __launch_bounds__(256) void k_s2(
    const _Float16* __restrict__ Cwh, const _Float16* __restrict__ Qh,
    const float* __restrict__ s0, const float* __restrict__ s1,
    const float* __restrict__ bias, const float* __restrict__ c_mask,
    const float* __restrict__ cmax, const float* __restrict__ csum,
    _Float16* __restrict__ S2h) {
    __shared__ float ls[32 * LSLD];
    const int b  = blockIdx.x / (CL / 32);
    const int ct = blockIdx.x % (CL / 32);
    const int c0 = ct * 32;

    compute_s_tile32(ls, b, c0, Cwh, Qh, s0, s1, bias[0]);
    __syncthreads();

    for (int i = threadIdx.x; i < 32 * QL; i += 256) {
        int r = i >> 9, q = i & (QL - 1);
        float cm = c_mask[b * CL + c0 + r];
        float sv = ls[r * LSLD + q] * cm + NEG_INF * (1.f - cm);
        float v  = __expf(sv - cmax[b * QL + q]) / csum[b * QL + q];
        S2h[(size_t)(b * CL + c0 + r) * QL + q] = (_Float16)v;
    }
}

// ---------------- fused main kernel ----------------
#define S1LD 520   // padded f16 row stride (1040B) for S1 and S2 LDS tiles
#define TLD  72    // padded f16 row stride for T tile

__global__ __launch_bounds__(256) void k_main(
    const _Float16* __restrict__ Cwh, const _Float16* __restrict__ Qh,
    const _Float16* __restrict__ QhT, const _Float16* __restrict__ ChT,
    const _Float16* __restrict__ S2h,
    const float* __restrict__ s0, const float* __restrict__ s1,
    const float* __restrict__ bias, const float* __restrict__ q_mask,
    const float* __restrict__ rmax, const float* __restrict__ rsum,
    const float* __restrict__ C, float* __restrict__ out) {
    __shared__ _Float16 s1l[64 * S1LD];       //  66,560 B : S1 tile (64 x 512)
    __shared__ _Float16 s2l[2 * 64 * S1LD];   // 133,120 B : S2 k-tile, dbl-buf (TDM)
    __shared__ _Float16 tst[2 * 64 * TLD];    //  18,432 B : T bounce, dbl-buf

    const int b    = blockIdx.x >> 5;         // 32 c-tiles of 64 per batch
    const int c0   = (blockIdx.x & 31) * 64;
    const int w    = threadIdx.x >> 5;
    const int lane = threadIdx.x & 31;
    const int n    = lane & 15;
    const int mb   = (lane >> 4) * 8;
    const int m    = w & 3;                   // M-tile (rows m*16..m*16+15)
    const int nb   = w >> 2;                  // 0/1 : 64-wide column block

    const unsigned s2ldsBase = (unsigned)(uintptr_t)(void*)s2l;
    const unsigned s2bufBytes = 64u * S1LD * 2u;

    // Prefetch S2 k-tile 0 via Tensor Data Mover (overlaps phases 1-2)
    if (w == 0) {
        tdm_load_s2_tile(S2h + (size_t)(b * CL) * QL, s2ldsBase);
    }

    // ---- Phase 1: S tile -> S1 = softmax_q(S) as f16 in LDS ----
    {
        const _Float16* Arow = Cwh + (size_t)(b * CL + c0 + m * 16) * DIM;
        v16h af[4];
#pragma unroll
        for (int kc = 0; kc < 4; ++kc) af[kc] = load_a_frag(Arow + kc * 32, DIM);
        float rmx[8], rsm[8], s0r[8];
#pragma unroll
        for (int r = 0; r < 8; ++r) {
            int cr = b * CL + c0 + m * 16 + mb + r;
            rmx[r] = rmax[cr]; rsm[r] = rsum[cr]; s0r[r] = s0[cr];
        }
        float bz = bias[0];
        for (int nt = 0; nt < 16; ++nt) {
            int q0 = nb * 256 + nt * 16;
            v8f acc = vzero8();
#pragma unroll
            for (int kc = 0; kc < 4; ++kc) {
                v16h bf = load_b_frag(Qh + (size_t)(b * QL + q0) * DIM + kc * 32, DIM);
                acc = wmma_f16(af[kc], bf, acc);
            }
            int   q   = q0 + n;
            float qm  = q_mask[b * QL + q];
            float s1v = s1[b * QL + q];
#pragma unroll
            for (int r = 0; r < 8; ++r) {
                float v  = acc[r] + s0r[r] + s1v + bz;
                float sv = v * qm + NEG_INF * (1.f - qm);
                float e  = __expf(sv - rmx[r]) / rsm[r];
                s1l[(m * 16 + mb + r) * S1LD + q] = (_Float16)e;
            }
        }
    }
    __syncthreads();

    // ---- Phase 2: A = S1 @ Q  (64 x 128, K = 512) ----
    v8f aacc[4] = {vzero8(), vzero8(), vzero8(), vzero8()};
    for (int kc = 0; kc < 16; ++kc) {
        v16h af = load_a_frag(&s1l[(m * 16) * S1LD + kc * 32], S1LD);
#pragma unroll
        for (int j = 0; j < 4; ++j) {
            v16h bf = load_b_frag(
                QhT + (size_t)(b * DIM + nb * 64 + j * 16) * QL + kc * 32, QL);
            aacc[j] = wmma_f16(af, bf, aacc[j]);
        }
    }

    // ---- Phase 3: Bm = (S1 @ S2^T) @ C, k-tiled (32 tiles of 64) ----
    v8f bmacc[4] = {vzero8(), vzero8(), vzero8(), vzero8()};
    for (int kt = 0; kt < 32; ++kt) {
        // Issue DMA for next tile, then wait for the current one (in-order cnt)
        if (w == 0) {
            if (kt + 1 < 32) {
                tdm_load_s2_tile(S2h + (size_t)(b * CL + (kt + 1) * 64) * QL,
                                 s2ldsBase + ((unsigned)(kt + 1) & 1u) * s2bufBytes);
                __builtin_amdgcn_s_wait_tensorcnt(1);
            } else {
                __builtin_amdgcn_s_wait_tensorcnt(0);
            }
        }
        __syncthreads();   // current S2 tile visible; prior tile reads all done

        const _Float16* s2buf = &s2l[(kt & 1) * 64 * S1LD];
        _Float16*       tbuf  = &tst[(kt & 1) * 64 * TLD];

        // T tile (64 x 64): this wave computes 1 M-tile x 2 N-tiles
        v8f tacc[2] = {vzero8(), vzero8()};
        for (int kc = 0; kc < 16; ++kc) {
            v16h af = load_a_frag(&s1l[(m * 16) * S1LD + kc * 32], S1LD);
#pragma unroll
            for (int j = 0; j < 2; ++j) {
                v16h bf = load_b_frag(&s2buf[((nb * 2 + j) * 16) * S1LD + kc * 32],
                                      S1LD);
                tacc[j] = wmma_f16(af, bf, tacc[j]);
            }
        }
        // D-layout -> f16 A-layout via LDS bounce (double-buffered)
#pragma unroll
        for (int j = 0; j < 2; ++j)
#pragma unroll
            for (int r = 0; r < 8; ++r)
                tbuf[(m * 16 + mb + r) * TLD + (nb * 2 + j) * 16 + n] =
                    (_Float16)tacc[j][r];
        __syncthreads();
        // Bm += T @ C[k-tile]   (K = 64)
#pragma unroll
        for (int kc2 = 0; kc2 < 2; ++kc2) {
            v16h af = load_a_frag(&tbuf[(m * 16) * TLD + kc2 * 32], TLD);
#pragma unroll
            for (int j = 0; j < 4; ++j) {
                v16h bf = load_b_frag(
                    ChT + (size_t)(b * DIM + nb * 64 + j * 16) * CL + kt * 64 + kc2 * 32, CL);
                bmacc[j] = wmma_f16(af, bf, bmacc[j]);
            }
        }
    }

    // ---- Phase 4: out = concat([C, A, C*A, C*Bm]) ----
#pragma unroll
    for (int j = 0; j < 4; ++j) {
        int dcol = nb * 64 + j * 16 + n;
#pragma unroll
        for (int r = 0; r < 8; ++r) {
            int    cg   = c0 + m * 16 + mb + r;
            float  cv   = C[(size_t)(b * CL + cg) * DIM + dcol];
            float  av   = aacc[j][r];
            float  bmv  = bmacc[j][r];
            size_t base = (size_t)(b * CL + cg) * 512;
            out[base + dcol]       = cv;
            out[base + 128 + dcol] = av;
            out[base + 256 + dcol] = cv * av;
            out[base + 384 + dcol] = cv * bmv;
        }
    }
}

// ---------------- launcher ----------------
extern "C" void kernel_launch(void* const* d_in, const int* in_sizes, int n_in,
                              void* d_out, int out_size, void* d_ws, size_t ws_size,
                              hipStream_t stream) {
    const float* C      = (const float*)d_in[0];
    const float* Q      = (const float*)d_in[1];
    const float* c_mask = (const float*)d_in[2];
    const float* q_mask = (const float*)d_in[3];
    const float* w4c    = (const float*)d_in[4];
    const float* w4q    = (const float*)d_in[5];
    const float* w4mul  = (const float*)d_in[6];
    const float* bias   = (const float*)d_in[7];

    char* ws = (char*)d_ws;
    size_t off = 0;
    auto take = [&](size_t bytes) {
        char* p = ws + off;
        off = (off + bytes + 255) & ~(size_t)255;
        return (void*)p;
    };
    _Float16* Cwh  = (_Float16*)take((size_t)BATCH * CL * DIM * 2);   // 32 MB
    _Float16* Qh   = (_Float16*)take((size_t)BATCH * QL * DIM * 2);   //  8 MB
    _Float16* QhT  = (_Float16*)take((size_t)BATCH * DIM * QL * 2);   //  8 MB
    _Float16* ChT  = (_Float16*)take((size_t)BATCH * DIM * CL * 2);   // 32 MB
    _Float16* S2h  = (_Float16*)take((size_t)BATCH * CL * QL * 2);    // 128 MB
    float* s0g  = (float*)take((size_t)BATCH * CL * 4);
    float* s1g  = (float*)take((size_t)BATCH * QL * 4);
    float* rmax = (float*)take((size_t)BATCH * CL * 4);
    float* rsum = (float*)take((size_t)BATCH * CL * 4);
    float* cmax = (float*)take((size_t)BATCH * QL * 4);
    float* csum = (float*)take((size_t)BATCH * QL * 4);
    float* pmax = (float*)take((size_t)BATCH * (CL / 32) * QL * 4);   //  8 MB
    float* psum = (float*)take((size_t)BATCH * (CL / 32) * QL * 4);   //  8 MB
    (void)in_sizes; (void)n_in; (void)out_size; (void)ws_size;

    k_prep_c<<<(BATCH * CL) / 256, 256, 0, stream>>>(C, w4c, w4mul, Cwh, ChT, s0g);
    k_prep_q<<<(BATCH * QL) / 256, 256, 0, stream>>>(Q, w4q, Qh, QhT, s1g);
    k_stats<<<BATCH * (CL / 32), 256, 0, stream>>>(Cwh, Qh, s0g, s1g, bias,
                                                   q_mask, c_mask,
                                                   rmax, rsum, pmax, psum);
    k_colred<<<(BATCH * QL) / 256, 256, 0, stream>>>(pmax, psum, cmax, csum);
    k_s2<<<BATCH * (CL / 32), 256, 0, stream>>>(Cwh, Qh, s0g, s1g, bias,
                                                c_mask, cmax, csum, S2h);
    k_main<<<BATCH * (CL / 64), 256, 0, stream>>>(Cwh, Qh, QhT, ChT, S2h,
                                                  s0g, s1g, bias, q_mask,
                                                  rmax, rsum, C, (float*)d_out);
}
